// GQAWithCache_10557029614340
// MI455X (gfx1250) — matrix operand order repaired and checked by reading, not compiled, not run on hardware
//
#include <hip/hip_runtime.h>

#define DEV __device__ __forceinline__

typedef __attribute__((ext_vector_type(16))) _Float16 v16h;
typedef __attribute__((ext_vector_type(8)))  _Float16 v8h;
typedef __attribute__((ext_vector_type(4)))  _Float16 v4h;
typedef __attribute__((ext_vector_type(8)))  float    v8f;
typedef __attribute__((ext_vector_type(4)))  float    v4f;
typedef __attribute__((ext_vector_type(4)))  int      v4i;

// pointer flavours for the async global->LDS builtin (AS1 = global, AS3 = LDS)
typedef __attribute__((address_space(1))) v4i gv4i;
typedef __attribute__((address_space(3))) v4i sv4i;

// problem constants (fixed by the reference)
constexpr int B_  = 2;
constexpr int S_  = 2048;
constexpr int P_  = 1024;
constexpr int T_  = 3072;   // P + S
constexpr int D_  = 4096;
constexpr int H_  = 32;
constexpr int KV_ = 8;
constexpr int HD_ = 128;
constexpr int OFF_ = T_ - S_;  // causal offset = 1024

// ---------------------------------------------------------------------------
// CDNA5 async global->LDS staging (ASYNCcnt-tracked), with safe fallback.
// ---------------------------------------------------------------------------
#if defined(__has_builtin)
#  if __has_builtin(__builtin_amdgcn_global_load_async_to_lds_b128)
#    define HAVE_ASYNC_LDS 1
#  endif
#endif
#ifndef HAVE_ASYNC_LDS
#  define HAVE_ASYNC_LDS 0
#endif

// copy 16 bytes (8 halfs) global -> LDS
DEV void cp16(_Float16* l, const _Float16* g) {
#if HAVE_ASYNC_LDS
  __builtin_amdgcn_global_load_async_to_lds_b128(
      (gv4i*)(void*)g, (sv4i*)(void*)l, 0, 0);
#else
  *(v8h*)l = *(const v8h*)g;
#endif
}

DEV void async_join() {
#if HAVE_ASYNC_LDS
#  if __has_builtin(__builtin_amdgcn_s_wait_asynccnt)
  __builtin_amdgcn_s_wait_asynccnt(0);
#  else
  asm volatile("s_wait_asynccnt 0" ::: "memory");
#  endif
#endif
}

// ---------------------------------------------------------------------------
// WMMA fragment loaders (16x16x32 f16, wave32; contiguous k = 16*(lane/16) + s order, not AMD's table order, same for A and B)
// A (16x32, MxK): lane 0-15 -> row M=lane, K 0..15; lane 16-31 -> same
// rows, K 16..31.  One contiguous 32B LDS load per lane.
// B (32x16, KxN): lane L -> column N = L&15; lanes 0-15 hold K 0..15, lanes
// 16-31 hold K 16..31, packed (2v,2v+1) per VGPR.  One contiguous 32B load
// when B is stored [n][k] in LDS.
// ---------------------------------------------------------------------------
DEV v16h load_a(const _Float16* p, int row0, int k0, int ldk, int lane) {
  const _Float16* q = p + (size_t)(row0 + (lane & 15)) * ldk + k0 + ((lane >> 4) << 4);
  return *(const v16h*)q;
}

DEV v16h load_b(const _Float16* p, int n0, int k0, int ldk, int lane) {
  const _Float16* q = p + (size_t)(n0 + (lane & 15)) * ldk + k0 + ((lane >> 4) << 4);
  return *(const v16h*)q;
}

DEV float hmax16(float v) {
  v = fmaxf(v, __shfl_xor(v, 1, 32));
  v = fmaxf(v, __shfl_xor(v, 2, 32));
  v = fmaxf(v, __shfl_xor(v, 4, 32));
  v = fmaxf(v, __shfl_xor(v, 8, 32));
  return v;
}
DEV float hsum16(float v) {
  v += __shfl_xor(v, 1, 32);
  v += __shfl_xor(v, 2, 32);
  v += __shfl_xor(v, 4, 32);
  v += __shfl_xor(v, 8, 32);
  return v;
}

// ---------------------------------------------------------------------------
// fp32 -> fp16 casts
// ---------------------------------------------------------------------------
__global__ void cast_f32_f16(const float* __restrict__ src,
                             _Float16* __restrict__ dst, int n4) {
  int i = blockIdx.x * blockDim.x + threadIdx.x;
  if (i < n4) {
    v4f f = ((const v4f*)src)[i];
    v4h h;
#pragma unroll
    for (int j = 0; j < 4; ++j) h[j] = (_Float16)f[j];
    ((v4h*)dst)[i] = h;
  }
}

// past_k/past_v [B,P,KV,HD] f32 -> rows [0,P) of [B,T,KV,HD] f16
__global__ void cast_cache(const float* __restrict__ src,
                           _Float16* __restrict__ dst, int n4) {
  int i = blockIdx.x * blockDim.x + threadIdx.x;
  if (i < n4) {
    v4f f = ((const v4f*)src)[i];
    v4h h;
#pragma unroll
    for (int j = 0; j < 4; ++j) h[j] = (_Float16)f[j];
    int elem = i * 4;
    int b    = elem >> 20;               // P*KV*HD = 2^20 elems per batch
    int rem  = elem & ((1 << 20) - 1);
    size_t d = (size_t)b * ((size_t)T_ * KV_ * HD_) + rem;
    *(v4h*)(dst + d) = h;
  }
}

// ---------------------------------------------------------------------------
// Tiled GEMM:  out[M x N] = A16[M x 4096] (f16) * W[4096 x N] (f32, cvt f16)
// Block 128x128, 8 waves (4m x 2n), each wave 32x64 = 2x4 16x16 frags.
// A tile staged with async global->LDS; W tile converted f32->f16 and stored
// transposed [n][k] so B-frags are contiguous 32B LDS loads.
// ---------------------------------------------------------------------------
template <bool OUT_F16>
__global__ __launch_bounds__(256) void gemm_f16wmma(
    const _Float16* __restrict__ A, const float* __restrict__ W,
    _Float16* __restrict__ outH, float* __restrict__ outF, int N,
    size_t outBStride, int outRowStride, int outColOff) {
  constexpr int K_ = 4096;
  __shared__ _Float16 Alds[128 * 32];   // [row][k]
  __shared__ _Float16 Blds[128 * 32];   // [n][k]  (transposed for B-frags)

  const int tid  = threadIdx.x;
  const int lane = tid & 31;
  const int wave = tid >> 5;
  const int wm   = wave >> 1;   // 0..3
  const int wn   = wave & 1;    // 0..1
  const int rowBase = blockIdx.y * 128;
  const int nBase   = blockIdx.x * 128;

  v8f acc[2][4];
  v8f zf = {};
#pragma unroll
  for (int i = 0; i < 2; ++i)
#pragma unroll
    for (int j = 0; j < 4; ++j) acc[i][j] = zf;

  for (int kt = 0; kt < K_ / 32; ++kt) {
    const int k0 = kt * 32;
    // stage A tile 128x32 f16: async global->LDS, 16B per lane per chunk
#pragma unroll
    for (int it = 0; it < 2; ++it) {
      int c = it * 256 + tid;
      int r = c >> 2, c8 = (c & 3) * 8;
      cp16(&Alds[r * 32 + c8], &A[(size_t)(rowBase + r) * K_ + k0 + c8]);
    }
    // stage W tile 32x128: f32 load, cvt f16, store transposed [n][k]
#pragma unroll
    for (int it = 0; it < 8; ++it) {
      int c  = it * 256 + tid;
      int n2 = (c & 63) * 2;
      int k  = c >> 6;
      const float* wp = &W[(size_t)(k0 + k) * N + nBase + n2];
      Blds[(n2)     * 32 + k] = (_Float16)wp[0];
      Blds[(n2 + 1) * 32 + k] = (_Float16)wp[1];
    }
    // prefetch next tiles into L2/WGP$ while this one is consumed
    if (kt + 1 < K_ / 32) {
      __builtin_prefetch(&W[(size_t)(k0 + 32 + (tid >> 3)) * N + nBase + (tid & 7) * 16], 0, 1);
      __builtin_prefetch(&A[(size_t)(rowBase + (tid >> 1)) * K_ + k0 + 32 + (tid & 1) * 16], 0, 1);
    }
    async_join();
    __syncthreads();

    v16h a0 = load_a(Alds, wm * 32 + 0,  0, 32, lane);
    v16h a1 = load_a(Alds, wm * 32 + 16, 0, 32, lane);
#pragma unroll
    for (int j = 0; j < 4; ++j) {
      v16h bfr = load_b(Blds, wn * 64 + j * 16, 0, 32, lane);
      acc[0][j] = __builtin_amdgcn_wmma_f32_16x16x32_f16(
          false, a0, false, bfr, (short)0, acc[0][j], false, false);
      acc[1][j] = __builtin_amdgcn_wmma_f32_16x16x32_f16(
          false, a1, false, bfr, (short)0, acc[1][j], false, false);
    }
    __syncthreads();
  }

  // store (C layout: VGPR e -> row e + 8*(lane>=16), col = lane&15)
  const int hi8 = (lane >> 4) * 8;
  const int nl  = lane & 15;
#pragma unroll
  for (int i = 0; i < 2; ++i)
#pragma unroll
    for (int j = 0; j < 4; ++j)
#pragma unroll
      for (int e = 0; e < 8; ++e) {
        int m  = rowBase + wm * 32 + i * 16 + e + hi8;
        int n  = nBase + wn * 64 + j * 16 + nl;
        int bb = m >> 11;         // S = 2048 rows per batch
        int ss = m & 2047;
        size_t idx = (size_t)bb * outBStride + (size_t)ss * outRowStride +
                     (size_t)(outColOff + n);
        if (OUT_F16)
          outH[idx] = (_Float16)acc[i][j][e];
        else
          outF[idx] = acc[i][j][e];
      }
}

// ---------------------------------------------------------------------------
// Flash attention (GQA, causal + cache offset) entirely on WMMA.
// Block = one (b, h) x 128 query rows; 8 waves x 16 rows each.
// Q/K tiles staged via async global->LDS; V transposed in-flight for B-frags.
// Key tiles past q+OFF skipped by the loop bound.
// ---------------------------------------------------------------------------
__global__ __launch_bounds__(256) void gqa_flash(
    const _Float16* __restrict__ q16, const _Float16* __restrict__ k16,
    const _Float16* __restrict__ v16, _Float16* __restrict__ o16) {
  extern __shared__ char smem_raw[];
  _Float16* Qlds = (_Float16*)smem_raw;   // [128][128]
  _Float16* Klds = Qlds + 128 * 128;      // [64][128]   row-major (t, d)
  _Float16* Vlds = Klds + 64 * 128;       // [128][64]   transposed (d, t)
  _Float16* Plds = Vlds + 128 * 64;       // [8 waves][16][64]

  const int tid  = threadIdx.x;
  const int lane = tid & 31;
  const int wave = tid >> 5;
  const int qt   = blockIdx.x;
  const int bh   = blockIdx.y;
  const int b    = bh >> 5;          // H = 32
  const int h    = bh & 31;
  const int kvh  = h >> 2;           // G = 4
  const int qbase = qt * 128;
  const int qrow0 = qbase + wave * 16;
  const int hi8   = (lane >> 4) * 8;
  const int nl    = lane & 15;

  const _Float16* qp = q16 + ((size_t)(b * S_ + qbase) * H_ + h) * HD_;
  const _Float16* kb = k16 + (size_t)b * T_ * KV_ * HD_ + (size_t)kvh * HD_;
  const _Float16* vb = v16 + (size_t)b * T_ * KV_ * HD_ + (size_t)kvh * HD_;

  // stage Q tile [128][128] (async)
#pragma unroll
  for (int it = 0; it < 8; ++it) {
    int c = it * 256 + tid;
    int r = c >> 4, c8 = (c & 15) * 8;
    cp16(&Qlds[r * 128 + c8], &qp[(size_t)r * (H_ * HD_) + c8]);
  }

  const float scale = 0.08838834764831845f;  // 1/sqrt(HD)
  float mrun[8], lrun[8];
  v8f o[8];
  v8f zf = {};
#pragma unroll
  for (int e = 0; e < 8; ++e) { mrun[e] = -3.0e38f; lrun[e] = 0.f; }
#pragma unroll
  for (int f = 0; f < 8; ++f) o[f] = zf;

  const int nkt = (qbase + 127 + OFF_) / 64 + 1;  // skip fully-masked tiles
  for (int tb = 0; tb < nkt; ++tb) {
    const int tbase = tb * 64;
    const _Float16* kp = kb + (size_t)tbase * (KV_ * HD_);
    const _Float16* vp = vb + (size_t)tbase * (KV_ * HD_);
    // K tile [64][128] f16 (async)
#pragma unroll
    for (int it = 0; it < 4; ++it) {
      int c = it * 256 + tid;
      int r = c >> 4, c8 = (c & 15) * 8;
      cp16(&Klds[r * 128 + c8], &kp[(size_t)r * (KV_ * HD_) + c8]);
    }
    // V tile transposed -> Vlds[d][t]
#pragma unroll
    for (int it = 0; it < 8; ++it) {
      int c = it * 256 + tid;
      int r = c >> 5, d4 = (c & 31) * 4;
      v4h x = *(const v4h*)&vp[(size_t)r * (KV_ * HD_) + d4];
#pragma unroll
      for (int u = 0; u < 4; ++u) Vlds[(d4 + u) * 64 + r] = x[u];
    }
    // prefetch next key/value tiles
    if (tb + 1 < nkt) {
      const _Float16* kn = kp + (size_t)64 * (KV_ * HD_);
      const _Float16* vn = vp + (size_t)64 * (KV_ * HD_);
      __builtin_prefetch(kn + tid * 32, 0, 1);
      __builtin_prefetch(vn + tid * 32, 0, 1);
    }
    async_join();
    __syncthreads();

    // S = Q K^T : 4 frags of 16x16, K-dim = HD in 4 steps of 32
    v8f sc[4];
#pragma unroll
    for (int j = 0; j < 4; ++j) sc[j] = zf;
#pragma unroll
    for (int kk = 0; kk < 128; kk += 32) {
      v16h a = load_a(Qlds, wave * 16, kk, 128, lane);
#pragma unroll
      for (int j = 0; j < 4; ++j) {
        v16h bfr = load_b(Klds, j * 16, kk, 128, lane);
        sc[j] = __builtin_amdgcn_wmma_f32_16x16x32_f16(
            false, a, false, bfr, (short)0, sc[j], false, false);
      }
    }

    const bool needMask = (tbase + 63) > (qrow0 + OFF_);
#pragma unroll
    for (int j = 0; j < 4; ++j)
#pragma unroll
      for (int e = 0; e < 8; ++e) {
        float v = sc[j][e] * scale;
        if (needMask) {
          int qi = qrow0 + e + hi8;
          int kj = tbase + j * 16 + nl;
          if (kj > qi + OFF_) v = -3.0e38f;
        }
        sc[j][e] = v;
      }

    // online softmax (row reductions across 16-lane half-groups)
    float alpha[8];
#pragma unroll
    for (int e = 0; e < 8; ++e) {
      float mx = fmaxf(fmaxf(sc[0][e], sc[1][e]), fmaxf(sc[2][e], sc[3][e]));
      mx = hmax16(mx);
      float mnew = fmaxf(mrun[e], mx);
      alpha[e] = __expf(mrun[e] - mnew);
      mrun[e]  = mnew;
    }
#pragma unroll
    for (int e = 0; e < 8; ++e) {
      float s = 0.f;
#pragma unroll
      for (int j = 0; j < 4; ++j) {
        float p = __expf(sc[j][e] - mrun[e]);
        sc[j][e] = p;
        s += p;
      }
      s = hsum16(s);
      lrun[e] = lrun[e] * alpha[e] + s;
    }
#pragma unroll
    for (int f = 0; f < 8; ++f)
#pragma unroll
      for (int e = 0; e < 8; ++e) o[f][e] *= alpha[e];

    // P: C-layout regs -> per-wave LDS in A-layout order
    _Float16* pw = Plds + wave * 1024;
#pragma unroll
    for (int j = 0; j < 4; ++j)
#pragma unroll
      for (int e = 0; e < 8; ++e)
        pw[(e + hi8) * 64 + j * 16 + nl] = (_Float16)sc[j][e];
    __syncthreads();

    // O += P V : 8 n-frags over HD, K-dim = 64 keys in 2 steps of 32
#pragma unroll
    for (int kk = 0; kk < 64; kk += 32) {
      v16h a = load_a(pw, 0, kk, 64, lane);
#pragma unroll
      for (int f = 0; f < 8; ++f) {
        v16h bfr = load_b(Vlds, f * 16, kk, 64, lane);
        o[f] = __builtin_amdgcn_wmma_f32_16x16x32_f16(
            false, a, false, bfr, (short)0, o[f], false, false);
      }
    }
    __syncthreads();
  }

  // epilogue: O /= l, store f16 into [B,S,H*HD]
  _Float16* op = o16 + (size_t)(b * S_ + qrow0) * D_ + (size_t)h * HD_;
#pragma unroll
  for (int e = 0; e < 8; ++e) {
    float inv = 1.0f / lrun[e];
    int m = e + hi8;
#pragma unroll
    for (int f = 0; f < 8; ++f)
      op[(size_t)m * D_ + f * 16 + nl] = (_Float16)(o[f][e] * inv);
  }
}

// ---------------------------------------------------------------------------
extern "C" void kernel_launch(void* const* d_in, const int* in_sizes, int n_in,
                              void* d_out, int out_size, void* d_ws,
                              size_t ws_size, hipStream_t stream) {
  (void)in_sizes; (void)n_in; (void)out_size; (void)ws_size;
  const float* x  = (const float*)d_in[0];
  const float* pk = (const float*)d_in[1];
  const float* pv = (const float*)d_in[2];
  const float* Wq = (const float*)d_in[3];
  const float* Wk = (const float*)d_in[4];
  const float* Wv = (const float*)d_in[5];
  const float* Wo = (const float*)d_in[6];
  float* out = (float*)d_out;

  // workspace layout (f16), total ~120 MB
  _Float16* ws  = (_Float16*)d_ws;
  _Float16* x16 = ws;                  // B*S*D        = 16777216
  _Float16* q16 = x16 + 16777216;      // B*S*H*HD     = 16777216
  _Float16* k16 = q16 + 16777216;      // B*T*KV*HD    =  6291456
  _Float16* v16 = k16 + 6291456;       // B*T*KV*HD    =  6291456
  _Float16* o16 = v16 + 6291456;       // B*S*D        = 16777216

  dim3 blk(256);
  cast_f32_f16<<<16384, blk, 0, stream>>>(x, x16, 16777216 / 4);
  cast_cache<<<2048, blk, 0, stream>>>(pk, k16, 2097152 / 4);
  cast_cache<<<2048, blk, 0, stream>>>(pv, v16, 2097152 / 4);

  // q = x@Wq  -> [B,S,H*HD]
  gemm_f16wmma<true><<<dim3(32, 32), blk, 0, stream>>>(
      x16, Wq, q16, nullptr, 4096, (size_t)S_ * 4096, 4096, 0);
  // k = x@Wk  -> rows [P,T) of [B,T,KV*HD]
  gemm_f16wmma<true><<<dim3(8, 32), blk, 0, stream>>>(
      x16, Wk, k16, nullptr, 1024, (size_t)T_ * KV_ * HD_, 1024, P_ * 1024);
  // v = x@Wv
  gemm_f16wmma<true><<<dim3(8, 32), blk, 0, stream>>>(
      x16, Wv, v16, nullptr, 1024, (size_t)T_ * KV_ * HD_, 1024, P_ * 1024);

  // flash attention: 16 query tiles x (B*H = 64) blocks, 80 KB dynamic LDS
  gqa_flash<<<dim3(16, 64), blk, 81920, stream>>>(q16, k16, v16, o16);

  // out = attn_out @ Wo (f32 result)
  gemm_f16wmma<false><<<dim3(32, 32), blk, 0, stream>>>(
      o16, Wo, nullptr, out, 4096, (size_t)S_ * 4096, 4096, 0);
}
